// MVSEM_61375082659879
// MI455X (gfx1250) — matrix operand order repaired and checked by reading, not compiled, not run on hardware
//
#include <hip/hip_runtime.h>
#include <hip/hip_bf16.h>
#include <stdint.h>

// ---------------------------------------------------------------------------
// Types for CDNA5 WMMA (wave32): v_wmma_f32_16x16x32_bf16
// ---------------------------------------------------------------------------
typedef __bf16 bf16;
typedef bf16  v16bf __attribute__((ext_vector_type(16)));
typedef bf16  v8bf  __attribute__((ext_vector_type(8)));
typedef float v8f   __attribute__((ext_vector_type(8)));

#define Bsz 32
#define Tsz 512
#define Hsz 512
#define G3  1536   // 3*H
#define ODIM 2048
#define Mtot (Bsz * Tsz)   // 16384
#define NWG_SCAN 16

// workspace offsets (bytes), all 256-aligned
static const size_t OFF_X    = 0;                              // 16384*512*2
static const size_t OFF_GX   = OFF_X    + (size_t)Mtot*Hsz*2;  // 16384*1536*4
static const size_t OFF_HS   = OFF_GX   + (size_t)Mtot*G3*4;   // 16384*512*2
static const size_t OFF_WIH  = OFF_HS   + (size_t)Mtot*Hsz*2;  // 1536*512*2
static const size_t OFF_WHH  = OFF_WIH  + (size_t)G3*Hsz*2;
static const size_t OFF_WPRJ = OFF_WHH  + (size_t)G3*Hsz*2;    // 2048*512*2
static const size_t OFF_HBUF = OFF_WPRJ + (size_t)ODIM*Hsz*2;  // 2*32*512*2
static const size_t OFF_CNT  = OFF_HBUF + (size_t)2*Bsz*Hsz*2;

__device__ __forceinline__ v8f v8f_zero() {
    v8f z = {0.f,0.f,0.f,0.f,0.f,0.f,0.f,0.f};
    return z;
}

__device__ __forceinline__ v8f wmma_bf16(v16bf a, v16bf b, v8f c) {
    // D = A(16x32) * B(32x16) + C, f32 accumulate
    return __builtin_amdgcn_wmma_f32_16x16x32_bf16(
        /*neg_a=*/false, a, /*neg_b=*/false, b,
        /*c_mod=*/(short)0, c, /*reuse_a=*/false, /*reuse_b=*/false);
}

// A operand: row-major [.. x ldk]; contiguous-K layout used here (B matches) for 16-bit 16x32 A:
// lanes 0-15 row M=lane, K = kk+0..15; lanes 16-31 K = kk+16..31
__device__ __forceinline__ v16bf load_a_tile(const bf16* base, int ldk,
                                             int rowBase, int kk, int lane) {
    int r  = rowBase + (lane & 15);
    int k0 = kk + ((lane >> 4) << 4);
    const bf16* p = base + (size_t)r * ldk + k0;
    v8bf lo = *(const v8bf*)(p);
    v8bf hi = *(const v8bf*)(p + 8);
    return __builtin_shufflevector(lo, hi, 0,1,2,3,4,5,6,7,8,9,10,11,12,13,14,15);
}

// B operand: weight stored row-major [N x ldk]; B[k][n] = W[n][k].
// Lane n=lane&15 holds 16 contiguous K: kk + (lane<16 ? 0 : 16)
__device__ __forceinline__ v16bf load_b_tile(const bf16* base, int ldk,
                                             int colBase, int kk, int lane) {
    int n  = colBase + (lane & 15);
    int k0 = kk + ((lane >> 4) << 4);
    return *(const v16bf*)(base + (size_t)n * ldk + k0);
}

// ---------------------------------------------------------------------------
// Prep kernels
// ---------------------------------------------------------------------------
__global__ void cast_f32_to_bf16(const float* __restrict__ s, bf16* __restrict__ d, int n) {
    for (int i = blockIdx.x * blockDim.x + threadIdx.x; i < n; i += gridDim.x * blockDim.x)
        d[i] = (bf16)s[i];
}

__global__ void embed_gather_bf16(const int* __restrict__ ids,
                                  const float* __restrict__ emb,
                                  bf16* __restrict__ xb) {
    int row = blockIdx.x;                  // b*T + t
    int id  = ids[row];
    const float* src = emb + (size_t)id * Hsz;
    bf16* dst = xb + (size_t)row * Hsz;
    for (int c = threadIdx.x; c < Hsz; c += blockDim.x)
        dst[c] = (id == 0) ? (bf16)0.f : (bf16)src[c];   // padding_idx = 0
}

__global__ void zero_counter(int* p) { *p = 0; }

// ---------------------------------------------------------------------------
// Big GEMM: C[M x N] = A[M x K](bf16) * W[N x K]^T(bf16) + bias, optional ReLU
// 8 waves/WG (wave32); wave tile 32x64 (2 Mtiles x 4 Ntiles); WG tile 64x256
// ---------------------------------------------------------------------------
template <bool RELU>
__global__ __launch_bounds__(256)
void gemm_bf16_wmma(const bf16* __restrict__ A, const bf16* __restrict__ W,
                    const float* __restrict__ bias, float* __restrict__ C,
                    int M, int N, int K) {
    const int lane = threadIdx.x & 31;
    const int wave = threadIdx.x >> 5;
    const int wr = wave >> 2;     // 0..1
    const int wc = wave & 3;      // 0..3
    const int rowBase = blockIdx.y * 64  + wr * 32;
    const int colBase = blockIdx.x * 256 + wc * 64;
    (void)M;

    v8f acc[2][4];
#pragma unroll
    for (int i = 0; i < 2; ++i)
#pragma unroll
        for (int j = 0; j < 4; ++j) acc[i][j] = v8f_zero();

    for (int kk = 0; kk < K; kk += 32) {
        if (kk + 32 < K) {   // pull next A rows toward L0/L2 early
            __builtin_prefetch(A + (size_t)(rowBase + (lane & 15)) * K + kk + 32, 0, 0);
        }
        v16bf a0 = load_a_tile(A, K, rowBase,      kk, lane);
        v16bf a1 = load_a_tile(A, K, rowBase + 16, kk, lane);
#pragma unroll
        for (int nt = 0; nt < 4; ++nt) {
            v16bf b = load_b_tile(W, K, colBase + nt * 16, kk, lane);
            acc[0][nt] = wmma_bf16(a0, b, acc[0][nt]);
            acc[1][nt] = wmma_bf16(a1, b, acc[1][nt]);
        }
    }
    // epilogue: D layout — lane&15 = n, lanes>=16 hold M+8, vgpr v = M row
#pragma unroll
    for (int mt = 0; mt < 2; ++mt) {
        int mrow = rowBase + mt * 16 + ((lane >> 4) << 3);
#pragma unroll
        for (int nt = 0; nt < 4; ++nt) {
            int n = colBase + nt * 16 + (lane & 15);
            float bs = bias[n];
#pragma unroll
            for (int v = 0; v < 8; ++v) {
                float val = acc[mt][nt][v] + bs;
                if (RELU) val = fmaxf(val, 0.f);
                C[(size_t)(mrow + v) * N + n] = val;
            }
        }
    }
}

// ---------------------------------------------------------------------------
// GRU recurrence: 16 persistent WGs, WG g owns hidden units [32g, 32g+32).
// Per step: async-stage full h into LDS (ASYNCcnt path, no VGPR round trip),
// 12 WMMA tiles (gh slice), gates, write h_new to double-buffered global h,
// grid barrier (atomic + s_sleep spin).
// ---------------------------------------------------------------------------
__device__ __forceinline__ void grid_sync(volatile int* counter, int& phase) {
    __syncthreads();
    __threadfence();
    ++phase;
    if (threadIdx.x == 0) {
        atomicAdd((int*)counter, 1);
        int target = NWG_SCAN * phase;
        while (*counter < target) __builtin_amdgcn_s_sleep(1);
    }
    __syncthreads();
}

__global__ __launch_bounds__(256)
void gru_scan(const float* __restrict__ gx,     // [B*T, 3H] (b*T+t major)
              const bf16* __restrict__ whh,     // [3H, H] bf16
              const float* __restrict__ bhh,    // [3H]
              bf16* __restrict__ hsb,           // [B*T, H] bf16 out
              bf16* __restrict__ hbuf,          // 2 x [B, H] bf16 (double buffer)
              volatile int* counter) {
    __shared__ bf16  hcur[Bsz * Hsz];   // 32 KB
    __shared__ float ghs[Bsz * 96];     // 12 KB: gh slice [batch][3*32]

    const int g     = blockIdx.x;
    const int jbase = g * 32;
    const int lane  = threadIdx.x & 31;
    const int wave  = threadIdx.x >> 5;
    int phase = 0;

    // LDS byte address of hcur (generic LDS addresses truncate to LDS offset)
    const uint32_t lds_hcur = (uint32_t)(uintptr_t)(void*)hcur;

    // h0 = 0 (each WG zeroes its own column slice of buffer 0)
    for (int e = threadIdx.x; e < Bsz * 32; e += 256) {
        int b = e >> 5, jl = e & 31;
        hbuf[(size_t)b * Hsz + jbase + jl] = (bf16)0.f;
    }
    grid_sync(counter, phase);

    for (int t = 0; t < Tsz; ++t) {
        // stage h(t) -> LDS via CDNA5 async global->LDS B128 copies (ASYNCcnt)
        {
            const char* hsrc = (const char*)(hbuf + (size_t)(t & 1) * (Bsz * Hsz));
            // 32KB / 16B = 2048 transfers; 8 per thread
            for (int i = threadIdx.x; i < (Bsz * Hsz * 2) / 16; i += 256) {
                uint32_t ldsa = lds_hcur + (uint32_t)i * 16u;
                const void* ga = (const void*)(hsrc + (size_t)i * 16);
                asm volatile("global_load_async_to_lds_b128 %0, %1, off"
                             :: "v"(ldsa), "v"(ga)
                             : "memory");
            }
            asm volatile("s_wait_asynccnt 0x0" ::: "memory");
        }
        __syncthreads();

        // gh slice = h(t) @ whh_slice^T : 2 Mtiles x 6 Ntiles = 12 tiles
        for (int tile = wave; tile < 12; tile += 8) {
            int mt = tile & 1;
            int nt = tile >> 1;          // 0..5
            int gate = nt >> 1;          // r/z/n
            int wrowBase = gate * Hsz + jbase + (nt & 1) * 16;
            v8f acc = v8f_zero();
            for (int kk = 0; kk < Hsz; kk += 32) {
                v16bf a = load_a_tile(hcur, Hsz, mt * 16, kk, lane);
                v16bf b = load_b_tile(whh, Hsz, wrowBase, kk, lane);
                acc = wmma_bf16(a, b, acc);
            }
            int col   = nt * 16 + (lane & 15);
            int rbase = mt * 16 + ((lane >> 4) << 3);
#pragma unroll
            for (int v = 0; v < 8; ++v) ghs[(rbase + v) * 96 + col] = acc[v];
        }
        __syncthreads();

        // gates + hidden update for this WG's 32 units x 32 batches
        for (int e = threadIdx.x; e < Bsz * 32; e += 256) {
            int b = e >> 5, jl = e & 31;
            int j = jbase + jl;
            const float* gxr = gx + ((size_t)b * Tsz + t) * G3;
            float ghr = ghs[b * 96 + jl]      + bhh[j];
            float ghz = ghs[b * 96 + 32 + jl] + bhh[Hsz + j];
            float ghn = ghs[b * 96 + 64 + jl] + bhh[2 * Hsz + j];
            float r = 1.f / (1.f + __expf(-(gxr[j] + ghr)));
            float z = 1.f / (1.f + __expf(-(gxr[Hsz + j] + ghz)));
            float n = tanhf(gxr[2 * Hsz + j] + r * ghn);
            float hold = (float)hcur[b * Hsz + j];
            float hnew = (1.f - z) * n + z * hold;
            bf16 hb = (bf16)hnew;
            hsb[((size_t)b * Tsz + t) * Hsz + j] = hb;
            hbuf[(size_t)((t + 1) & 1) * (Bsz * Hsz) + (size_t)b * Hsz + j] = hb;
        }
        grid_sync(counter, phase);
    }
}

// ---------------------------------------------------------------------------
// Host launcher
// ---------------------------------------------------------------------------
extern "C" void kernel_launch(void* const* d_in, const int* in_sizes, int n_in,
                              void* d_out, int out_size, void* d_ws, size_t ws_size,
                              hipStream_t stream) {
    (void)in_sizes; (void)n_in; (void)out_size; (void)ws_size;
    const int*   ids    = (const int*)  d_in[0];
    const float* emb    = (const float*)d_in[1];
    const float* w_ih   = (const float*)d_in[2];
    const float* w_hh   = (const float*)d_in[3];
    const float* b_ih   = (const float*)d_in[4];
    const float* b_hh   = (const float*)d_in[5];
    const float* w_proj = (const float*)d_in[6];
    const float* b_proj = (const float*)d_in[7];
    float* out = (float*)d_out;

    char* ws = (char*)d_ws;
    bf16*  xb    = (bf16*) (ws + OFF_X);
    float* gx    = (float*)(ws + OFF_GX);
    bf16*  hsb   = (bf16*) (ws + OFF_HS);
    bf16*  wihb  = (bf16*) (ws + OFF_WIH);
    bf16*  whhb  = (bf16*) (ws + OFF_WHH);
    bf16*  wprjb = (bf16*) (ws + OFF_WPRJ);
    bf16*  hbuf  = (bf16*) (ws + OFF_HBUF);
    int*   cnt   = (int*)  (ws + OFF_CNT);

    // 1) weight casts f32 -> bf16
    cast_f32_to_bf16<<<512, 256, 0, stream>>>(w_ih,   wihb,  G3 * Hsz);
    cast_f32_to_bf16<<<512, 256, 0, stream>>>(w_hh,   whhb,  G3 * Hsz);
    cast_f32_to_bf16<<<512, 256, 0, stream>>>(w_proj, wprjb, ODIM * Hsz);

    // 2) embedding gather (+padding_idx=0) -> bf16
    embed_gather_bf16<<<Mtot, 128, 0, stream>>>(ids, emb, xb);

    // 3) reset grid-sync counter (deterministic across replays)
    zero_counter<<<1, 1, 0, stream>>>(cnt);

    // 4) gx = x @ w_ih^T + b_ih   [16384 x 1536], K=512
    gemm_bf16_wmma<false><<<dim3(G3 / 256, Mtot / 64), 256, 0, stream>>>(
        xb, wihb, b_ih, gx, Mtot, G3, Hsz);

    // 5) sequential GRU scan (persistent 16-WG grid, WMMA per step)
    gru_scan<<<NWG_SCAN, 256, 0, stream>>>(gx, whhb, b_hh, hsb, hbuf, cnt);

    // 6) out = relu(hs @ w_proj^T + b_proj)   [16384 x 2048], K=512
    gemm_bf16_wmma<true><<<dim3(ODIM / 256, Mtot / 64), 256, 0, stream>>>(
        hsb, wprjb, b_proj, out, Mtot, ODIM, Hsz);
}
